// SNN_14980845929059
// MI455X (gfx1250) — compile-verified, matrix operand order checked by reading, not run on hardware
//
#include <hip/hip_runtime.h>

// ---------------- problem constants ----------------
#define T_STEPS 500
#define B_SZ    64
#define IN_D    512
#define HID_D   2048
#define OUT_D   128
#define MROWS   (B_SZ * T_STEPS)     // 32000 rows of the batched GEMMs

// layer-1 scan constants
#define A_S1  0.81873075307798182f
#define SC1   0.90634623461009088f
#define A_M1  0.90483741803595952f
#define OM_M1 0.09516258196404048f
#define TH1   0.5f
// layer-2 scan constants
#define A_S2  0.90483741803595952f
#define SC2   0.95162581964040482f
#define A_M2  0.95122942450071400f
#define OM_M2 0.04877057549928600f
#define TH2   1.0f

typedef __bf16 bf16x16 __attribute__((ext_vector_type(16)));
typedef float  f32x8   __attribute__((ext_vector_type(8)));
typedef int    v4i_t   __attribute__((vector_size(16)));

#define AS1 __attribute__((address_space(1)))
#define AS3 __attribute__((address_space(3)))

union Frag {                 // 32 bytes = one WMMA bf16 A/B fragment per lane
    bf16x16 v;
    uint4   q[2];
};

// gfx1250 async global->LDS copy (device pass only; host pass takes fallback)
#if defined(__gfx1250__) && __has_builtin(__builtin_amdgcn_global_load_async_to_lds_b128)
#define ASYNC_LDS 1
#else
#define ASYNC_LDS 0
#endif

__device__ __forceinline__ unsigned short f2bf(float f) {
    union { float f; unsigned u; } x; x.f = f;
    unsigned r = x.u + 0x7FFFu + ((x.u >> 16) & 1u);   // round-to-nearest-even
    return (unsigned short)(r >> 16);
}

// ---------------- conversion / packing ----------------
__global__ void snn_cvt_bf16(const float* __restrict__ src,
                             unsigned short* __restrict__ dst, int n) {
    int i = blockIdx.x * blockDim.x + threadIdx.x;
    if (i < n) dst[i] = f2bf(src[i]);
}

// Pack row-major [K x N] fp32 weights into the per-lane WMMA B layout:
// element index i = ((ntile*ksteps + kk)*32 + lane)*16 + e
// lane: n = ntile*16 + (lane&15), kb = (lane>>4)*8
// e<8  -> K = kk*32 + kb + e ; e>=8 -> K = kk*32 + 16 + kb + (e-8)
__global__ void snn_pack_w(const float* __restrict__ w,
                           unsigned short* __restrict__ wp,
                           int ksteps, int ncols, int total) {
    int i = blockIdx.x * blockDim.x + threadIdx.x;
    if (i >= total) return;
    int e    = i & 15;
    int lane = (i >> 4) & 31;
    int kk   = (i >> 9) % ksteps;
    int nt   = i / (ksteps << 9);
    int kb   = (lane >> 4) << 3;
    int kl   = (e < 8) ? (kb + e) : (16 + kb + (e - 8));
    int K    = kk * 32 + kl;
    int n    = nt * 16 + (lane & 15);
    wp[i] = f2bf(w[(size_t)K * ncols + n]);
}

// ---------------- batched GEMM: C[M x N] = A[M x K] * Bpacked ----------------
// Block = 256 threads = 8 waves. Block computes 8 M-tiles x 64 cols (4 N-tiles).
// All 8 waves share the same 4 B tiles: staged through double-buffered LDS
// (async global->LDS when available), one 4 KB stage per k-step.
__global__ void __launch_bounds__(256)
snn_gemm_wmma(const unsigned short* __restrict__ Abf,  // row-major bf16
              const unsigned short* __restrict__ Bp,   // packed layout
              float* __restrict__ C,
              int lda, int ldc, int ksteps,
              int nblocks, int mtiles) {
    __shared__ __align__(16) unsigned short bsm[2][4 * 512];   // 2 x 4KB

    const int tid  = threadIdx.x;
    const int lane = tid & 31;
    const int wv   = tid >> 5;
    const int mb   = blockIdx.x / nblocks;
    const int nb   = blockIdx.x % nblocks;
    int mt = mb * 8 + wv;
    if (mt >= mtiles) mt = mtiles - 1;      // stay barrier-safe

    // cooperative B copy: thread copies 16B; chunk (tile) = tid>>6, slot = tid&63
    const int cn = tid >> 6, ci = tid & 63;
    const unsigned short* bsrc0 =
        Bp + (((size_t)(nb * 4 + cn) * ksteps) << 9) + ci * 8;

    const int m  = mt * 16 + (lane & 15);
    const int kb = (lane >> 4) << 3;                    // 0 or 8
    const unsigned short* arow = Abf + (size_t)m * lda;

    f32x8 acc[4] = {};

#if ASYNC_LDS
    {
        unsigned short* l = &bsm[0][cn * 512 + ci * 8]; // kk = 0
        __builtin_amdgcn_global_load_async_to_lds_b128(
            (AS1 v4i_t*)bsrc0, (AS3 v4i_t*)l, 0, 0);
    }
    for (int kk = 0; kk < ksteps; ++kk) {
        asm volatile("s_wait_asynccnt 0x0" ::: "memory");
        __syncthreads();                                // buf[kk&1] ready for all
        if (kk + 1 < ksteps) {                          // prefetch next k-step
            const unsigned short* g = bsrc0 + ((size_t)(kk + 1) << 9);
            unsigned short* l = &bsm[(kk + 1) & 1][cn * 512 + ci * 8];
            __builtin_amdgcn_global_load_async_to_lds_b128(
                (AS1 v4i_t*)g, (AS3 v4i_t*)l, 0, 0);
        }
        Frag a;
        a.q[0] = *(const uint4*)(arow + kk * 32 + kb);
        a.q[1] = *(const uint4*)(arow + kk * 32 + 16 + kb);
        const unsigned short* bbuf = bsm[kk & 1];
        // load all 4 B fragments into distinct regs, then issue WMMAs
        Frag b0, b1, b2, b3;
        b0.q[0] = *(const uint4*)(bbuf + 0 * 512 + lane * 16);
        b0.q[1] = *(const uint4*)(bbuf + 0 * 512 + lane * 16 + 8);
        b1.q[0] = *(const uint4*)(bbuf + 1 * 512 + lane * 16);
        b1.q[1] = *(const uint4*)(bbuf + 1 * 512 + lane * 16 + 8);
        b2.q[0] = *(const uint4*)(bbuf + 2 * 512 + lane * 16);
        b2.q[1] = *(const uint4*)(bbuf + 2 * 512 + lane * 16 + 8);
        b3.q[0] = *(const uint4*)(bbuf + 3 * 512 + lane * 16);
        b3.q[1] = *(const uint4*)(bbuf + 3 * 512 + lane * 16 + 8);
        acc[0] = __builtin_amdgcn_wmma_f32_16x16x32_bf16(
            false, a.v, false, b0.v, (short)0, acc[0], false, false);
        acc[1] = __builtin_amdgcn_wmma_f32_16x16x32_bf16(
            false, a.v, false, b1.v, (short)0, acc[1], false, false);
        acc[2] = __builtin_amdgcn_wmma_f32_16x16x32_bf16(
            false, a.v, false, b2.v, (short)0, acc[2], false, false);
        acc[3] = __builtin_amdgcn_wmma_f32_16x16x32_bf16(
            false, a.v, false, b3.v, (short)0, acc[3], false, false);
    }
#else
    uint4 stage = *(const uint4*)(bsrc0);               // kk = 0
    for (int kk = 0; kk < ksteps; ++kk) {
        *(uint4*)(&bsm[kk & 1][cn * 512 + ci * 8]) = stage;
        __syncthreads();                                // stores visible
        if (kk + 1 < ksteps)                            // overlap with compute
            stage = *(const uint4*)(bsrc0 + ((size_t)(kk + 1) << 9));
        Frag a;
        a.q[0] = *(const uint4*)(arow + kk * 32 + kb);
        a.q[1] = *(const uint4*)(arow + kk * 32 + 16 + kb);
        const unsigned short* bbuf = bsm[kk & 1];
        Frag b0, b1, b2, b3;
        b0.q[0] = *(const uint4*)(bbuf + 0 * 512 + lane * 16);
        b0.q[1] = *(const uint4*)(bbuf + 0 * 512 + lane * 16 + 8);
        b1.q[0] = *(const uint4*)(bbuf + 1 * 512 + lane * 16);
        b1.q[1] = *(const uint4*)(bbuf + 1 * 512 + lane * 16 + 8);
        b2.q[0] = *(const uint4*)(bbuf + 2 * 512 + lane * 16);
        b2.q[1] = *(const uint4*)(bbuf + 2 * 512 + lane * 16 + 8);
        b3.q[0] = *(const uint4*)(bbuf + 3 * 512 + lane * 16);
        b3.q[1] = *(const uint4*)(bbuf + 3 * 512 + lane * 16 + 8);
        acc[0] = __builtin_amdgcn_wmma_f32_16x16x32_bf16(
            false, a.v, false, b0.v, (short)0, acc[0], false, false);
        acc[1] = __builtin_amdgcn_wmma_f32_16x16x32_bf16(
            false, a.v, false, b1.v, (short)0, acc[1], false, false);
        acc[2] = __builtin_amdgcn_wmma_f32_16x16x32_bf16(
            false, a.v, false, b2.v, (short)0, acc[2], false, false);
        acc[3] = __builtin_amdgcn_wmma_f32_16x16x32_bf16(
            false, a.v, false, b3.v, (short)0, acc[3], false, false);
        __syncthreads();                                // reads done before overwrite
    }
#endif

    // C/D layout: VGPR r -> M = r + 8*(lane>=16), N = lane&15
    const int row0 = mt * 16 + ((lane >> 4) << 3);
    const int col0 = lane & 15;
#pragma unroll
    for (int n = 0; n < 4; ++n) {
        const int col = (nb * 4 + n) * 16 + col0;
#pragma unroll
        for (int r = 0; r < 8; ++r)
            C[(size_t)(row0 + r) * ldc + col] = acc[n][r];
    }
}

// ---------------- layer-1 temporal scan ----------------
__global__ void snn_scan1(const float* __restrict__ z1,
                          float* __restrict__ spk1,
                          float* __restrict__ mems1,
                          unsigned short* __restrict__ s1b) {
    int idx = blockIdx.x * blockDim.x + threadIdx.x;      // 0 .. B*HID-1
    if (idx >= B_SZ * HID_D) return;
    int b = idx / HID_D, h = idx % HID_D;
    size_t base = (size_t)b * T_STEPS * HID_D + h;
    float syn = 0.f, mem = 0.f;
    for (int t = 0; t < T_STEPS; ++t) {
        size_t o = base + (size_t)t * HID_D;
        float z = z1[o];
        syn = A_S1 * syn + SC1 * z;
        mem = A_M1 * mem + OM_M1 * syn;
        float s = (mem - TH1) > 0.f ? 1.f : 0.f;
        mem -= s * TH1;
        spk1[o]  = s;
        mems1[o] = mem;
        s1b[o]   = (s > 0.f) ? (unsigned short)0x3F80u : (unsigned short)0u;
    }
}

// ---------------- layer-2 temporal scan + output ----------------
__global__ void snn_scan2(const float* __restrict__ z2,
                          float* __restrict__ out,
                          float* __restrict__ spk2,
                          float* __restrict__ mems2) {
    int idx = blockIdx.x * blockDim.x + threadIdx.x;      // 0 .. B*OUT-1
    if (idx >= B_SZ * OUT_D) return;
    int b = idx / OUT_D, o = idx % OUT_D;
    size_t base = (size_t)b * T_STEPS * OUT_D + o;
    float syn = 0.f, mem = 0.f, sum = 0.f;
    for (int t = 0; t < T_STEPS; ++t) {
        size_t ofs = base + (size_t)t * OUT_D;
        float z = z2[ofs];
        syn = A_S2 * syn + SC2 * z;
        mem = A_M2 * mem + OM_M2 * syn;
        float s = (mem - TH2) > 0.f ? 1.f : 0.f;
        mem -= s * TH2;
        spk2[ofs]  = s;
        mems2[ofs] = mem;
        if (t > 0) sum += mem;
    }
    out[idx] = sum * (1.0f / (float)T_STEPS);
}

// ---------------- host side ----------------
extern "C" void kernel_launch(void* const* d_in, const int* in_sizes, int n_in,
                              void* d_out, int out_size, void* d_ws, size_t ws_size,
                              hipStream_t stream) {
    const float* x  = (const float*)d_in[0];   // [B,T,IN]
    const float* w1 = (const float*)d_in[1];   // [IN,HID]
    const float* w2 = (const float*)d_in[2];   // [HID,OUT]
    float* out = (float*)d_out;

    // output slices, concatenated flat in return order
    const size_t n_spk1 = (size_t)B_SZ * T_STEPS * HID_D;
    const size_t n_spk2 = (size_t)B_SZ * T_STEPS * OUT_D;
    float* o_out   = out;
    float* o_spk1  = out + (size_t)B_SZ * OUT_D;
    float* o_mems1 = o_spk1 + n_spk1;
    float* o_spk2  = o_mems1 + n_spk1;
    float* o_mems2 = o_spk2 + n_spk2;

    // workspace partition (256B aligned)
    char* ws = (char*)d_ws;
    size_t off = 0;
    auto carve = [&](size_t bytes) -> void* {
        void* p = ws + off;
        off += (bytes + 255) & ~(size_t)255;
        return p;
    };
    const int NX  = B_SZ * T_STEPS * IN_D;          // 16,384,000
    const int NW1 = IN_D * HID_D;                   //  1,048,576
    const int NW2 = HID_D * OUT_D;                  //    262,144
    unsigned short* xb  = (unsigned short*)carve((size_t)NX * 2);
    unsigned short* w1p = (unsigned short*)carve((size_t)NW1 * 2);
    unsigned short* w2p = (unsigned short*)carve((size_t)NW2 * 2);
    float*          z1  = (float*)carve((size_t)MROWS * HID_D * 4);
    unsigned short* s1b = (unsigned short*)carve((size_t)MROWS * HID_D * 2);
    float*          z2  = (float*)carve((size_t)MROWS * OUT_D * 4);
    (void)ws_size; (void)in_sizes; (void)n_in; (void)out_size;

    // 1) convert activations + pack weights to bf16 WMMA layouts
    snn_cvt_bf16<<<(NX + 255) / 256, 256, 0, stream>>>(x, xb, NX);
    snn_pack_w<<<(NW1 + 255) / 256, 256, 0, stream>>>(w1, w1p, IN_D / 32, HID_D, NW1);
    snn_pack_w<<<(NW2 + 255) / 256, 256, 0, stream>>>(w2, w2p, HID_D / 32, OUT_D, NW2);

    // 2) Z1 = X @ W1  : M=32000 (250 m-blocks of 8 tiles), K=512, N=2048 (32 n-blocks)
    {
        const int mtiles = MROWS / 16, nblocks = HID_D / 64;
        const int mblocks = (mtiles + 7) / 8;        // 250
        snn_gemm_wmma<<<mblocks * nblocks, 256, 0, stream>>>(
            xb, w1p, z1, IN_D, HID_D, IN_D / 32, nblocks, mtiles);
    }

    // 3) layer-1 scan
    snn_scan1<<<(B_SZ * HID_D) / 256, 256, 0, stream>>>(z1, o_spk1, o_mems1, s1b);

    // 4) Z2 = S1 @ W2 : M=32000, K=2048, N=128 (2 n-blocks)
    {
        const int mtiles = MROWS / 16, nblocks = OUT_D / 64;
        const int mblocks = (mtiles + 7) / 8;        // 250
        snn_gemm_wmma<<<mblocks * nblocks, 256, 0, stream>>>(
            s1b, w2p, z2, HID_D, OUT_D, HID_D / 32, nblocks, mtiles);
    }

    // 5) layer-2 scan + averaged output
    snn_scan2<<<(B_SZ * OUT_D) / 256, 256, 0, stream>>>(z2, o_out, o_spk2, o_mems2);
}
